// ReversiGNN_14748917694868
// MI455X (gfx1250) — matrix operand order repaired and checked by reading, not compiled, not run on hardware
//
#include <hip/hip_runtime.h>

// ---------- WMMA vector types ----------
typedef _Float16 half8  __attribute__((ext_vector_type(8)));
typedef _Float16 v16h   __attribute__((ext_vector_type(16)));
typedef float    v8f    __attribute__((ext_vector_type(8)));

#define H 64

// ---------- 1. deg[n] = 1 (self loop) ----------
__global__ void k_init_deg(float* __restrict__ deg, int n) {
    int i = blockIdx.x * blockDim.x + threadIdx.x;
    if (i < n) deg[i] = 1.0f;
}

// ---------- 2. deg[dst] += 1 per edge ----------
__global__ void k_deg_edges(const int* __restrict__ ei, float* __restrict__ deg, int e) {
    int i = blockIdx.x * blockDim.x + threadIdx.x;
    if (i < e) atomicAdd(&deg[ei[e + i]], 1.0f);
}

// ---------- 3. dinv = rsqrt(deg); s = x*dinv^2 (self-loop term) ----------
__global__ void k_dinv_s(const float* __restrict__ deg, const float* __restrict__ x,
                         float* __restrict__ dinv, float* __restrict__ s, int n) {
    int i = blockIdx.x * blockDim.x + threadIdx.x;
    if (i < n) {
        float d  = deg[i];
        float di = d > 0.0f ? rsqrtf(d) : 0.0f;
        dinv[i] = di;
        s[i]    = x[i] * di * di;
    }
}

// ---------- 4. layer-1 scalar edge scatter: s[dst] += x[src]*dinv[src]*dinv[dst] ----------
__global__ void k_agg1(const int* __restrict__ ei, const float* __restrict__ x,
                       const float* __restrict__ dinv, float* __restrict__ s, int e) {
    int i = blockIdx.x * blockDim.x + threadIdx.x;
    if (i < e) {
        int src = ei[i];
        int dst = ei[e + i];
        atomicAdd(&s[dst], x[src] * dinv[src] * dinv[dst]);
    }
}

// ---------- 5. h1 = relu(s ⊗ w1 + b1), stored f16 for WMMA ----------
__global__ void k_h1(const float* __restrict__ s, const float* __restrict__ w1,
                     const float* __restrict__ b1, _Float16* __restrict__ h1, int n) {
    int i = blockIdx.x * blockDim.x + threadIdx.x;
    if (i < n * H) {
        int node = i >> 6, j = i & 63;
        float v = fmaxf(s[node] * w1[j] + b1[j], 0.0f);
        h1[i] = (_Float16)v;
    }
}

// ---------- 6. w2^T in f16 (B fragments become contiguous b128 loads) ----------
__global__ void k_w2t(const float* __restrict__ w2, _Float16* __restrict__ w2t) {
    int i = blockIdx.x * blockDim.x + threadIdx.x;   // i = j*64 + k
    if (i < H * H) {
        int j = i >> 6, k = i & 63;
        w2t[i] = (_Float16)w2[k * H + j];
    }
}

// ---------- 7. t = h1 @ w2 via v_wmma_f32_16x16x32_f16, one 16x16 tile per wave ----------
__global__ void __launch_bounds__(256)
k_gemm(const _Float16* __restrict__ h1, const _Float16* __restrict__ w2t,
       float* __restrict__ t, int n) {
    int wave = (blockIdx.x * blockDim.x + threadIdx.x) >> 5;
    int lane = threadIdx.x & 31;
    int mt = wave >> 2;          // M tile (16 rows)
    int nt = wave & 3;           // N tile (4 tiles of 16 cols)
    if (mt * 16 >= n) return;    // wave-uniform -> EXEC all-1s for WMMA

    int l15  = lane & 15;
    int half = lane >> 4;
    int row  = mt * 16 + l15;    // A row for this lane
    int col  = nt * 16 + l15;    // B column / D column for this lane

    v8f c = {};
    #pragma unroll
    for (int ko = 0; ko < 64; ko += 32) {
        union { v16h v; half8 h[2]; } a, b;
        // A 16x32 f16 layout: VGPR0-3 -> K = ko + half*8 + [0,8); VGPR4-7 -> +16
        const _Float16* ap = h1 + row * H + ko + half * 8;
        a.h[0] = *(const half8*)(ap);
        a.h[1] = *(const half8*)(ap + 16);
        // B 32x16 f16 layout: lane = column, VGPR0-3 -> K = ko + half*8 + [0,8); VGPR4-7 -> +16
        const _Float16* bp = w2t + col * H + ko + half * 8;
        b.h[0] = *(const half8*)(bp);
        b.h[1] = *(const half8*)(bp + 16);
        c = __builtin_amdgcn_wmma_f32_16x16x32_f16(
                /*neg_a=*/false, a.v, /*neg_b=*/false, b.v,
                /*c_mod=*/(short)0, c, /*reuse_a=*/false, /*reuse_b=*/false);
    }
    // D 16x16 f32: lane col = l15, rows = half*8 + r
    int mbase = mt * 16 + half * 8;
    #pragma unroll
    for (int r = 0; r < 8; ++r)
        t[(mbase + r) * H + col] = c[r];
}

// ---------- 8. agg2 = t * dinv^2 (self-loop init) ----------
__global__ void k_agg2_init(const float* __restrict__ t, const float* __restrict__ dinv,
                            float* __restrict__ agg2, int n) {
    int i = blockIdx.x * blockDim.x + threadIdx.x;
    if (i < n * H) {
        int node = i >> 6;
        float di = dinv[node];
        agg2[i] = t[i] * di * di;
    }
}

// ---------- 9. layer-2 edge scatter: agg2[dst] += t[src]*norm, 16 lanes/edge, float4 ----------
__global__ void k_agg2_edges(const int* __restrict__ ei, const float* __restrict__ dinv,
                             const float* __restrict__ t, float* __restrict__ agg2, int e) {
    long long gid = (long long)blockIdx.x * blockDim.x + threadIdx.x;
    long long tot = (long long)e * 16;
    if (gid >= tot) return;
    int ed = (int)(gid >> 4);
    int q  = (int)(gid & 15);
    int src = ei[ed];
    int dst = ei[e + ed];
    float norm = dinv[src] * dinv[dst];
    const float4 v = *(const float4*)(t + (size_t)src * H + q * 4);
    float* out = agg2 + (size_t)dst * H + q * 4;
    atomicAdd(out + 0, v.x * norm);
    atomicAdd(out + 1, v.y * norm);
    atomicAdd(out + 2, v.z * norm);
    atomicAdd(out + 3, v.w * norm);
}

// ---------- 10. out = relu(agg2 + b2) @ wfc + bfc, one wave32 per node ----------
__global__ void __launch_bounds__(256)
k_out(const float* __restrict__ agg2, const float* __restrict__ b2,
      const float* __restrict__ wfc, const float* __restrict__ bfc,
      float* __restrict__ out, int n) {
    int node = blockIdx.x * 8 + (threadIdx.x >> 5);
    int lane = threadIdx.x & 31;
    if (node >= n) return;
    const float* row = agg2 + (size_t)node * H;
    float acc = fmaxf(row[lane]      + b2[lane],      0.0f) * wfc[lane]
              + fmaxf(row[lane + 32] + b2[lane + 32], 0.0f) * wfc[lane + 32];
    #pragma unroll
    for (int o = 16; o > 0; o >>= 1)
        acc += __shfl_xor(acc, o, 32);
    if (lane == 0) out[node] = acc + bfc[0];
}

extern "C" void kernel_launch(void* const* d_in, const int* in_sizes, int n_in,
                              void* d_out, int out_size, void* d_ws, size_t ws_size,
                              hipStream_t stream) {
    const float* x   = (const float*)d_in[0];
    const int*   ei  = (const int*)  d_in[1];
    const float* w1  = (const float*)d_in[2];
    const float* b1  = (const float*)d_in[3];
    const float* w2  = (const float*)d_in[4];
    const float* b2  = (const float*)d_in[5];
    const float* wfc = (const float*)d_in[6];
    const float* bfc = (const float*)d_in[7];
    float* out = (float*)d_out;

    const int N = in_sizes[0];        // 100000
    const int E = in_sizes[1] / 2;    // 1600000

    // workspace carve-out (256B aligned)
    char* ws = (char*)d_ws;
    size_t off = 0;
    auto take = [&](size_t bytes) { char* p = ws + off; off = (off + bytes + 255) & ~(size_t)255; return p; };
    float*    deg   = (float*)   take((size_t)N * 4);
    float*    dinv  = (float*)   take((size_t)N * 4);
    float*    s     = (float*)   take((size_t)N * 4);
    _Float16* h1    = (_Float16*)take((size_t)N * H * 2);
    _Float16* w2t   = (_Float16*)take((size_t)H * H * 2);
    float*    t     = (float*)   take((size_t)N * H * 4);
    float*    agg2  = (float*)   take((size_t)N * H * 4);
    (void)ws_size; (void)n_in; (void)out_size;

    const int TB = 256;
    auto cdiv = [](long long a, long long b) { return (int)((a + b - 1) / b); };

    k_w2t      <<<cdiv(H * H, TB),          TB, 0, stream>>>(w2, w2t);
    k_init_deg <<<cdiv(N, TB),              TB, 0, stream>>>(deg, N);
    k_deg_edges<<<cdiv(E, TB),              TB, 0, stream>>>(ei, deg, E);
    k_dinv_s   <<<cdiv(N, TB),              TB, 0, stream>>>(deg, x, dinv, s, N);
    k_agg1     <<<cdiv(E, TB),              TB, 0, stream>>>(ei, x, dinv, s, E);
    k_h1       <<<cdiv((long long)N * H, TB), TB, 0, stream>>>(s, w1, b1, h1, N);

    int waves  = ((N + 15) / 16) * 4;       // 16-row M tiles x 4 N tiles
    int blocks = (waves + 7) / 8;           // 8 waves per 256-thread block
    k_gemm     <<<blocks,                   TB, 0, stream>>>(h1, w2t, t, N);

    k_agg2_init <<<cdiv((long long)N * H, TB),  TB, 0, stream>>>(t, dinv, agg2, N);
    k_agg2_edges<<<cdiv((long long)E * 16, TB), TB, 0, stream>>>(ei, dinv, t, agg2, E);
    k_out       <<<cdiv(N, 8),                  TB, 0, stream>>>(agg2, b2, wfc, bfc, out, N);
}
